// TransformerCritic_18769007083820
// MI455X (gfx1250) — compile-verified, matrix operand order checked by reading, not run on hardware
//
#include <hip/hip_runtime.h>
#include <hip/hip_bf16.h>
#include <cmath>

// ---------------------------------------------------------------------------
// TransformerCritic forward for MI455X (gfx1250, wave32, WMMA).
// Strategy: all linear layers as f16 WMMA GEMMs (v_wmma_f32_16x16x32_f16) with
// fused bias/GELU/residual/LayerNorm epilogues; fully fused GRU kernel with
// in-register WMMA recurrence + fused value head. Everything stays L2-resident.
// ---------------------------------------------------------------------------

typedef __attribute__((ext_vector_type(16))) _Float16 v16h;
typedef __attribute__((ext_vector_type(8)))  float    v8f;

#define DEV __device__ __forceinline__

// WMMA fragment loader. 16-bit A-matrix layout (16x32, MxK), per CDNA5 ISA:
//   lanes 0-15 : M = lane,  VGPR j<4 -> K = 2j,2j+1 ; j>=4 -> K = 16+2(j-4)..
//   lanes 16-31: same M,    K offset +8 (j<4) / +8 in upper half (j>=4)
// B-matrix (32x16, KxN) uses the mirrored layout with N striped across lanes,
// so the same loader works with `base` = W + ntile*16*Kp (W stored N x K).
DEV v16h frag_load(const _Float16* __restrict__ base, int Kp, int k0, int lane) {
  const int mr  = lane & 15;
  const int grp = lane >> 4;
  const _Float16* p = base + (size_t)mr * Kp + k0 + grp * 8;
  v16h f;
#pragma unroll
  for (int j = 0; j < 8; ++j) {
    const int kb = (j < 4) ? (2 * j) : (16 + 2 * (j - 4));
    f[2 * j]     = p[kb];
    f[2 * j + 1] = p[kb + 1];
  }
  return f;
}

DEV v8f wmma_f16(v16h a, v16h b, v8f c) {
  return __builtin_amdgcn_wmma_f32_16x16x32_f16(false, a, false, b, (short)0, c,
                                                false, false);
}

DEV float gelu_exact(float x) { return 0.5f * x * (1.0f + erff(x * 0.70710678118654752f)); }
DEV float sigmoidf(float x)   { return 1.0f / (1.0f + __expf(-x)); }

enum { F_GELU = 1, F_LN = 2, F_RES = 4, F_PERM = 8 };

// ---------------------------------------------------------------------------
// Generic GEMM: out = epilogue(A(MxKp,f16) @ W(64xKp,f16)^T + bias)
// One wave -> one 16x64 tile (4 WMMA accumulators). Epilogue via LDS.
// F_PERM remaps output rows (bt,i,j) -> (n=(b,i,j), t) for the GRU x-layout.
// ---------------------------------------------------------------------------
__global__ __launch_bounds__(128) void gemm64_kernel(
    const _Float16* __restrict__ A, const _Float16* __restrict__ W,
    const float* __restrict__ bias, const float* __restrict__ lng,
    const float* __restrict__ lnb, const _Float16* __restrict__ res,
    _Float16* __restrict__ out, int M, int Kp, int flags) {
  __shared__ float smem[4][16][64];
  const int lane = threadIdx.x & 31;
  const int wv   = threadIdx.x >> 5;
  const int m0   = (blockIdx.x * 4 + wv) * 16;
  if (m0 < M) {
    v8f acc0 = {}, acc1 = {}, acc2 = {}, acc3 = {};
    const _Float16* Arow = A + (size_t)m0 * Kp;
    for (int k0 = 0; k0 < Kp; k0 += 32) {
      __builtin_prefetch(Arow + (size_t)(lane & 15) * Kp + k0 + 64, 0, 1);
      v16h a  = frag_load(Arow, Kp, k0, lane);
      v16h b0 = frag_load(W + (size_t)0 * 16 * Kp, Kp, k0, lane);
      v16h b1 = frag_load(W + (size_t)1 * 16 * Kp, Kp, k0, lane);
      v16h b2 = frag_load(W + (size_t)2 * 16 * Kp, Kp, k0, lane);
      v16h b3 = frag_load(W + (size_t)3 * 16 * Kp, Kp, k0, lane);
      acc0 = wmma_f16(a, b0, acc0);
      acc1 = wmma_f16(a, b1, acc1);
      acc2 = wmma_f16(a, b2, acc2);
      acc3 = wmma_f16(a, b3, acc3);
    }
    const int nlo = lane & 15, grp = lane >> 4;
    const float bc0 = bias[nlo], bc1 = bias[16 + nlo], bc2 = bias[32 + nlo],
                bc3 = bias[48 + nlo];
#pragma unroll
    for (int r = 0; r < 8; ++r) {
      const int mr = r + 8 * grp;   // C-fragment row mapping
      smem[wv][mr][nlo]      = acc0[r] + bc0;
      smem[wv][mr][16 + nlo] = acc1[r] + bc1;
      smem[wv][mr][32 + nlo] = acc2[r] + bc2;
      smem[wv][mr][48 + nlo] = acc3[r] + bc3;
    }
  }
  __syncthreads();
  const int t = threadIdx.x;
  if (t < 64) {
    const int gm = blockIdx.x * 64 + t;
    if (gm < M) {
      const int sw = t >> 4, sm = t & 15;
      float v[64];
#pragma unroll
      for (int c = 0; c < 64; ++c) v[c] = smem[sw][sm][c];
      if (flags & F_GELU) {
#pragma unroll
        for (int c = 0; c < 64; ++c) v[c] = gelu_exact(v[c]);
      }
      if (flags & F_RES) {
#pragma unroll
        for (int c = 0; c < 64; ++c) v[c] += (float)res[(size_t)gm * 64 + c];
      }
      if (flags & F_LN) {
        float mu = 0.f;
#pragma unroll
        for (int c = 0; c < 64; ++c) mu += v[c];
        mu *= (1.f / 64.f);
        float var = 0.f;
#pragma unroll
        for (int c = 0; c < 64; ++c) { float d = v[c] - mu; var += d * d; }
        var *= (1.f / 64.f);
        const float rs = rsqrtf(var + 1e-5f);
#pragma unroll
        for (int c = 0; c < 64; ++c) v[c] = (v[c] - mu) * rs * lng[c] + lnb[c];
      }
      size_t orow = (size_t)gm;
      if (flags & F_PERM) {  // (bt,i,j) with bt=b*50+t  ->  ((b,i,j)*50 + t)
        const int bt = gm / 100, rem = gm - bt * 100;
        const int ii = rem / 10, jj = rem - ii * 10;
        const int b = bt / 50, tt = bt - b * 50;
        orow = (size_t)(((b * 10 + ii) * 10 + jj) * 50 + tt);
      }
#pragma unroll
      for (int c = 0; c < 64; ++c) out[orow * 64 + c] = (_Float16)v[c];
    }
  }
}

// GRU input precompute: gi = x @ W_ih^T + b_ih  (M x 64 -> M x 192), f16 out.
__global__ __launch_bounds__(128) void gemm192_kernel(
    const _Float16* __restrict__ A, const _Float16* __restrict__ W,
    const float* __restrict__ bias, _Float16* __restrict__ out, int M) {
  const int lane = threadIdx.x & 31;
  const int wv   = threadIdx.x >> 5;
  const int m0   = (blockIdx.x * 4 + wv) * 16;
  if (m0 >= M) return;
  const _Float16* Arow = A + (size_t)m0 * 64;
  const v16h a0 = frag_load(Arow, 64, 0, lane);
  const v16h a1 = frag_load(Arow, 64, 32, lane);
  const int nlo = lane & 15, grp = lane >> 4;
#pragma unroll
  for (int nt = 0; nt < 12; ++nt) {
    v16h b0 = frag_load(W + (size_t)nt * 16 * 64, 64, 0, lane);
    v16h b1 = frag_load(W + (size_t)nt * 16 * 64, 64, 32, lane);
    v8f acc = {};
    acc = wmma_f16(a0, b0, acc);
    acc = wmma_f16(a1, b1, acc);
    const float bc = bias[nt * 16 + nlo];
#pragma unroll
    for (int r = 0; r < 8; ++r)
      out[(size_t)(m0 + r + 8 * grp) * 192 + nt * 16 + nlo] =
          (_Float16)(acc[r] + bc);
  }
}

// ---------------------------------------------------------------------------
// Fused GRU + value head. One wave owns 16 of the 1600 independent sequences.
// h lives in LDS; recurrence h@W_hh^T (64->192) = 24 WMMAs per step in-register.
// Value head (LN + dot) fused per timestep; hT written at the end.
// ---------------------------------------------------------------------------
__global__ __launch_bounds__(32) void gru_kernel(
    const _Float16* __restrict__ gi, const _Float16* __restrict__ Whh,
    const float* __restrict__ bhh, const float* __restrict__ h0,
    const float* __restrict__ vg, const float* __restrict__ vb,
    const float* __restrict__ vw, const float* __restrict__ vbias,
    float* __restrict__ outv) {
  __shared__ float h[16][64];
  const int lane = threadIdx.x;
  const int n0   = blockIdx.x * 16;
  for (int idx = lane; idx < 1024; idx += 32) {
    const int m = idx >> 6, d = idx & 63;
    h[m][d] = h0[(size_t)(n0 + m) * 64 + d];
  }
  __syncthreads();
  const int nlo = lane & 15, grp = lane >> 4;
  for (int t = 0; t < 50; ++t) {
    // A fragments from h (f32 LDS -> f16), 16x64 split into two 16x32 frags
    v16h a0, a1;
#pragma unroll
    for (int j = 0; j < 8; ++j) {
      const int kb = ((j < 4) ? (2 * j) : (16 + 2 * (j - 4))) + grp * 8;
      a0[2 * j]     = (_Float16)h[nlo][kb];
      a0[2 * j + 1] = (_Float16)h[nlo][kb + 1];
      a1[2 * j]     = (_Float16)h[nlo][32 + kb];
      a1[2 * j + 1] = (_Float16)h[nlo][32 + kb + 1];
    }
    v8f acc[12];
#pragma unroll
    for (int nt = 0; nt < 12; ++nt) {
      v16h b0 = frag_load(Whh + (size_t)nt * 16 * 64, 64, 0, lane);
      v16h b1 = frag_load(Whh + (size_t)nt * 16 * 64, 64, 32, lane);
      v8f c = {};
      c = wmma_f16(a0, b0, c);
      acc[nt] = wmma_f16(a1, b1, c);
    }
    __syncthreads();  // all lanes done reading h before updating it
#pragma unroll
    for (int t4 = 0; t4 < 4; ++t4) {
      const int c0 = t4 * 16 + nlo;
#pragma unroll
      for (int r = 0; r < 8; ++r) {
        const int m = r + 8 * grp;
        const int n = n0 + m;
        const size_t gir = (size_t)(n * 50 + t) * 192;
        const float gr  = acc[t4][r]     + bhh[c0]       + (float)gi[gir + c0];
        const float gz  = acc[t4 + 4][r] + bhh[c0 + 64]  + (float)gi[gir + c0 + 64];
        const float ghn = acc[t4 + 8][r] + bhh[c0 + 128];
        const float gin = (float)gi[gir + c0 + 128];
        const float rr = sigmoidf(gr);
        const float zz = sigmoidf(gz);
        const float nn = tanhf(gin + rr * ghn);
        h[m][c0] = (1.f - zz) * nn + zz * h[m][c0];
      }
    }
    __syncthreads();
    if (lane < 16) {  // fused value head: LN(h) . vw + vbias
      const int m = lane, n = n0 + m;
      float mu = 0.f;
      for (int d = 0; d < 64; ++d) mu += h[m][d];
      mu *= (1.f / 64.f);
      float var = 0.f;
      for (int d = 0; d < 64; ++d) { float dd = h[m][d] - mu; var += dd * dd; }
      var *= (1.f / 64.f);
      const float rs = rsqrtf(var + 1e-5f);
      float av = vbias[0];
      for (int d = 0; d < 64; ++d)
        av += ((h[m][d] - mu) * rs * vg[d] + vb[d]) * vw[d];
      const int b = n / 100, rem = n - b * 100;
      const int ii = rem / 10, jj = rem - ii * 10;
      outv[(size_t)((b * 50 + t) * 10 + ii) * 10 + jj] = av;
    }
    __syncthreads();
  }
  for (int idx = lane; idx < 1024; idx += 32) {  // hT output
    const int m = idx >> 6, d = idx & 63;
    outv[400000 + (size_t)(n0 + m) * 64 + d] = h[m][d];
  }
}

// --------------------------- glue / staging kernels ------------------------

// f32 (N,K) -> f16 (N,Kp) zero-padded (weights + activation staging)
__global__ void cvt_pad_kernel(const float* __restrict__ src,
                               _Float16* __restrict__ dst, int N, int K, int Kp) {
  const int i = blockIdx.x * blockDim.x + threadIdx.x;
  if (i >= N * Kp) return;
  const int n = i / Kp, k = i - n * Kp;
  dst[i] = (k < K) ? (_Float16)src[(size_t)n * K + k] : (_Float16)0.f;
}

// oap[(bt,i,j), 0:192] = concat(states[bt,j,:172], (i==j?pol:act)[bt,j,:14], 0pad)
__global__ void build_oap_kernel(const float* __restrict__ states,
                                 const float* __restrict__ actions,
                                 const float* __restrict__ policies,
                                 _Float16* __restrict__ oap) {
  const int i = blockIdx.x * blockDim.x + threadIdx.x;
  if (i >= 80000 * 192) return;
  const int row = i / 192, k = i - row * 192;
  const int bt = row / 100, rem = row - bt * 100;
  const int ai = rem / 10, aj = rem - ai * 10;
  float v = 0.f;
  if (k < 172)
    v = states[(size_t)(bt * 10 + aj) * 172 + k];
  else if (k < 186) {
    const float* src = (ai == aj) ? policies : actions;
    v = src[(size_t)(bt * 10 + aj) * 14 + (k - 172)];
  }
  oap[i] = (_Float16)v;
}

// 10x10 masked multi-head softmax; writes ret_weight directly into d_out.
__global__ __launch_bounds__(64) void ally_attn_kernel(
    const _Float16* __restrict__ q, const _Float16* __restrict__ k,
    const float* __restrict__ masks, float* __restrict__ wout) {
  const int bt = blockIdx.x;
  const int t  = threadIdx.x;
  if (t >= 40) return;
  const int h = t / 10, qi = t - h * 10;
  const float mi = masks[bt * 10 + qi];
  float s[10];
  float mx = -3.0e38f;
  const _Float16* qp = q + (size_t)(bt * 10 + qi) * 64 + h * 16;
  for (int j = 0; j < 10; ++j) {
    const _Float16* kp = k + (size_t)(bt * 10 + j) * 64 + h * 16;
    float dot = 0.f;
#pragma unroll
    for (int d = 0; d < 16; ++d) dot += (float)qp[d] * (float)kp[d];
    const float mj = masks[bt * 10 + j];
    const float sc = dot * 0.25f + (((mi == 0.f) || (mj == 0.f)) ? -3.4e38f : 0.f);
    s[j] = sc;
    mx = fmaxf(mx, sc);
  }
  float den = 0.f;
  for (int j = 0; j < 10; ++j) { s[j] = __expf(s[j] - mx); den += s[j]; }
  const float inv = 1.f / den;
  float* wrow = wout + (size_t)((bt * 4 + h) * 10 + qi) * 10;
  for (int j = 0; j < 10; ++j)
    wrow[j] = s[j] * inv * mi * masks[bt * 10 + j];
}

// nf[bt,x,y,c] = LN( oape[bt,x,y,c] + sum_k av[bt,y,k,c] * w[bt,c/16,x,k] )
__global__ __launch_bounds__(64) void nf_kernel(
    const _Float16* __restrict__ av, const float* __restrict__ w,
    const _Float16* __restrict__ oape, const float* __restrict__ lng,
    const float* __restrict__ lnb, _Float16* __restrict__ nf) {
  __shared__ float wsh[4][10];
  __shared__ float red[64];
  const int bx = blockIdx.x;  // bt*10 + x
  const int bt = bx / 10, x = bx - bt * 10;
  const int c = threadIdx.x;
  const int h = c >> 4;
  if (c < 40) {
    const int hh = c / 10, kk = c - hh * 10;
    wsh[hh][kk] = w[(size_t)((bt * 4 + hh) * 10 + x) * 10 + kk];
  }
  __syncthreads();
  for (int y = 0; y < 10; ++y) {
    float acc = 0.f;
#pragma unroll
    for (int kk = 0; kk < 10; ++kk)
      acc += (float)av[(size_t)((bt * 10 + y) * 10 + kk) * 64 + c] * wsh[h][kk];
    const float val =
        acc + (float)oape[(size_t)((bt * 10 + x) * 10 + y) * 64 + c];
    red[c] = val;
    __syncthreads();
    float mu = 0.f;
    for (int i2 = 0; i2 < 64; ++i2) mu += red[i2];
    mu *= (1.f / 64.f);
    __syncthreads();
    const float dv = val - mu;
    red[c] = dv * dv;
    __syncthreads();
    float var = 0.f;
    for (int i2 = 0; i2 < 64; ++i2) var += red[i2];
    var *= (1.f / 64.f);
    __syncthreads();
    const float rs = rsqrtf(var + 1e-5f);
    nf[(size_t)((bt * 10 + x) * 10 + y) * 64 + c] =
        (_Float16)(dv * rs * lng[c] + lnb[c]);
  }
}

// agg_base = LN( mean_e(ee) + se + softmax(qe.ke/8) @ ve ) over 11 enemies
__global__ __launch_bounds__(64) void enemy_kernel(
    const _Float16* __restrict__ qe, const _Float16* __restrict__ ke,
    const _Float16* __restrict__ ve, const _Float16* __restrict__ ee,
    const _Float16* __restrict__ se, const float* __restrict__ lng,
    const float* __restrict__ lnb, _Float16* __restrict__ agg) {
  __shared__ float wesh[10][11];
  __shared__ float red[64];
  const int bt = blockIdx.x;
  const int t  = threadIdx.x;
  if (t < 10) {
    float s[11];
    float mx = -3.0e38f;
    const _Float16* qp = qe + (size_t)(bt * 10 + t) * 64;
    for (int e = 0; e < 11; ++e) {
      const _Float16* kp = ke + (size_t)(bt * 11 + e) * 64;
      float dot = 0.f;
      for (int d = 0; d < 64; ++d) dot += (float)qp[d] * (float)kp[d];
      s[e] = dot * 0.125f;
      mx = fmaxf(mx, s[e]);
    }
    float den = 0.f;
    for (int e = 0; e < 11; ++e) { s[e] = __expf(s[e] - mx); den += s[e]; }
    const float inv = 1.f / den;
    for (int e = 0; e < 11; ++e) wesh[t][e] = s[e] * inv;
  }
  __syncthreads();
  const int d = t;
  float meanee = 0.f;
  for (int e = 0; e < 11; ++e) meanee += (float)ee[(size_t)(bt * 11 + e) * 64 + d];
  meanee *= (1.f / 11.f);
  for (int a = 0; a < 10; ++a) {
    float v = meanee + (float)se[(size_t)(bt * 10 + a) * 64 + d];
    for (int e = 0; e < 11; ++e)
      v += wesh[a][e] * (float)ve[(size_t)(bt * 11 + e) * 64 + d];
    red[d] = v;
    __syncthreads();
    float mu = 0.f;
    for (int i2 = 0; i2 < 64; ++i2) mu += red[i2];
    mu *= (1.f / 64.f);
    __syncthreads();
    const float dv = v - mu;
    red[d] = dv * dv;
    __syncthreads();
    float var = 0.f;
    for (int i2 = 0; i2 < 64; ++i2) var += red[i2];
    var *= (1.f / 64.f);
    __syncthreads();
    const float rs = rsqrtf(var + 1e-5f);
    agg[(size_t)(bt * 10 + a) * 64 + d] = (_Float16)(dv * rs * lng[d] + lnb[d]);
  }
}

// cur_in[(bt,i,j),0:192] = concat(se[bt,i], agg[bt,j], nf[bt,i,j])
__global__ void concat_kernel(const _Float16* __restrict__ se,
                              const _Float16* __restrict__ agg,
                              const _Float16* __restrict__ nf,
                              _Float16* __restrict__ cur) {
  const int i = blockIdx.x * blockDim.x + threadIdx.x;
  if (i >= 80000 * 192) return;
  const int row = i / 192, kk = i - row * 192;
  const int bt = row / 100, rem = row - bt * 100;
  const int ai = rem / 10, aj = rem - ai * 10;
  _Float16 v;
  if (kk < 64)        v = se[(size_t)(bt * 10 + ai) * 64 + kk];
  else if (kk < 128)  v = agg[(size_t)(bt * 10 + aj) * 64 + (kk - 64)];
  else                v = nf[(size_t)row * 64 + (kk - 128)];
  cur[i] = v;
}

// ---------------------------------------------------------------------------
// Host orchestration. Param indices assume setup_inputs() dict insertion-order
// recursive flattening: states, enemy_states, policies, actions,
// rnn_hidden_state, agent_masks, then params leaves depth-first.
// ---------------------------------------------------------------------------
extern "C" void kernel_launch(void* const* d_in, const int* in_sizes, int n_in,
                              void* d_out, int out_size, void* d_ws,
                              size_t ws_size, hipStream_t stream) {
  (void)in_sizes; (void)n_in; (void)out_size; (void)ws_size;

  const float* states   = (const float*)d_in[0];
  const float* enemies  = (const float*)d_in[1];
  const float* policies = (const float*)d_in[2];
  const float* actions  = (const float*)d_in[3];
  const float* h0       = (const float*)d_in[4];
  const float* masks    = (const float*)d_in[5];
  auto P = [&](int i) -> const float* { return (const float*)d_in[i]; };
  // ally_embed
  const float *ae_w = P(6), *ae_b = P(7), *ae_g = P(8), *ae_lb = P(9);
  // sap_embed
  const float *sap_w = P(10), *sap_b = P(11), *sap_g = P(12), *sap_lb = P(13);
  const float *key_w = P(14), *key_b = P(15), *qry_w = P(16), *qry_b = P(17);
  const float *val_w = P(18), *val_b = P(19);
  const float *avln_g = P(20), *avln_b = P(21);
  const float *l1_w = P(22), *l1_b = P(23), *l1g = P(24), *l1lb = P(25);
  const float *l2_w = P(26), *l2_b = P(27);
  const float *avlinln_g = P(28), *avlinln_b = P(29);
  const float *eew = P(30), *eeb = P(31), *eeg = P(32), *eelb = P(33);
  const float *ke_w = P(34), *ke_b = P(35), *qe_w = P(36), *qe_b = P(37);
  const float *ve_w = P(38), *ve_b = P(39);
  const float *aveln_g = P(40), *aveln_b = P(41);
  const float *e1_w = P(42), *e1_b = P(43), *e1g = P(44), *e1lb = P(45);
  const float *e2_w = P(46), *e2_b = P(47);
  const float *avelnln_g = P(48), *avelnln_b = P(49);
  const float *com_w = P(50), *com_b = P(51);
  const float *wih = P(52), *whh = P(53), *bih = P(54), *bhh_ = P(55);
  const float *vg = P(56), *vbp = P(57), *vw = P(58), *vbias = P(59);

  float* outv = (float*)d_out;          // value   [0, 80000)
  float* wbuf = outv + 80000;           // weights [80000, 400000) (also reused)
  // hT written at outv+400000 inside gru_kernel.

  // ---- workspace arena (f16, 256B aligned) ----
  char* base = (char*)d_ws;
  size_t off = 0;
  auto A16 = [&](size_t elems) -> _Float16* {
    off = (off + 255) & ~(size_t)255;
    _Float16* p = (_Float16*)(base + off);
    off += elems * sizeof(_Float16);
    return p;
  };
  _Float16 *Wae = A16(64 * 192), *Wsap = A16(64 * 192), *Wq = A16(64 * 64),
           *Wk = A16(64 * 64), *Wv = A16(64 * 64), *Wl1 = A16(64 * 64),
           *Wl2 = A16(64 * 64), *Wee = A16(64 * 128), *Wqe = A16(64 * 64),
           *Wke = A16(64 * 64), *Wve = A16(64 * 64), *We1 = A16(64 * 64),
           *We2 = A16(64 * 64), *Wcom = A16(64 * 192), *Wih = A16(192 * 64),
           *Whh = A16(192 * 64);
  _Float16 *s16 = A16((size_t)8000 * 192), *es16 = A16((size_t)8800 * 128);
  _Float16 *oap16 = A16((size_t)80000 * 192);  // reused later as cur_in
  _Float16 *se16 = A16((size_t)8000 * 64), *q16 = A16((size_t)8000 * 64),
           *k16 = A16((size_t)8000 * 64), *qe16 = A16((size_t)8000 * 64);
  _Float16 *ee16 = A16((size_t)8800 * 64), *ke16 = A16((size_t)8800 * 64),
           *ve16 = A16((size_t)8800 * 64);
  _Float16 *oape16 = A16((size_t)80000 * 64), *av16 = A16((size_t)80000 * 64),
           *nf16 = A16((size_t)80000 * 64), *nff16 = A16((size_t)80000 * 64);
  _Float16 *aggb16 = A16((size_t)8000 * 64), *t2_16 = A16((size_t)8000 * 64),
           *agg16 = A16((size_t)8000 * 64);
  _Float16 *curx16 = A16((size_t)80000 * 64), *gi16 = A16((size_t)80000 * 192);
  _Float16* curin16 = oap16;  // oap dead after sap_embed GEMM
  _Float16* t1_16   = av16;   // av dead after nf einsum

  auto cvt = [&](const float* src, _Float16* dst, int N, int K, int Kp) {
    const int total = N * Kp;
    cvt_pad_kernel<<<dim3((total + 255) / 256), dim3(256), 0, stream>>>(
        src, dst, N, K, Kp);
  };
  auto g64 = [&](const _Float16* Ain, const _Float16* Wm, const float* bias,
                 const float* lg, const float* lb, const _Float16* res,
                 _Float16* outp, int M, int Kp, int flags) {
    gemm64_kernel<<<dim3((M + 63) / 64), dim3(128), 0, stream>>>(
        Ain, Wm, bias, lg, lb, res, outp, M, Kp, flags);
  };

  // ---- weight conversion (f32 -> f16, K zero-padded to /32) ----
  cvt(ae_w, Wae, 64, 172, 192);
  cvt(sap_w, Wsap, 64, 186, 192);
  cvt(qry_w, Wq, 64, 64, 64);
  cvt(key_w, Wk, 64, 64, 64);
  cvt(val_w, Wv, 64, 64, 64);
  cvt(l1_w, Wl1, 64, 64, 64);
  cvt(l2_w, Wl2, 64, 64, 64);
  cvt(eew, Wee, 64, 112, 128);
  cvt(qe_w, Wqe, 64, 64, 64);
  cvt(ke_w, Wke, 64, 64, 64);
  cvt(ve_w, Wve, 64, 64, 64);
  cvt(e1_w, We1, 64, 64, 64);
  cvt(e2_w, We2, 64, 64, 64);
  cvt(com_w, Wcom, 64, 192, 192);
  cvt(wih, Wih, 192, 64, 64);
  cvt(whh, Whh, 192, 64, 64);

  // ---- activation staging ----
  cvt(states, s16, 8000, 172, 192);
  cvt(enemies, es16, 8800, 112, 128);
  build_oap_kernel<<<dim3(60000), dim3(256), 0, stream>>>(states, actions,
                                                          policies, oap16);

  // ---- ally branch ----
  g64(s16, Wae, ae_b, ae_g, ae_lb, nullptr, se16, 8000, 192, F_GELU | F_LN);
  g64(oap16, Wsap, sap_b, sap_g, sap_lb, nullptr, oape16, 80000, 192,
      F_GELU | F_LN);
  g64(se16, Wq, qry_b, nullptr, nullptr, nullptr, q16, 8000, 64, 0);
  g64(se16, Wk, key_b, nullptr, nullptr, nullptr, k16, 8000, 64, 0);
  ally_attn_kernel<<<dim3(800), dim3(64), 0, stream>>>(q16, k16, masks, wbuf);
  g64(oape16, Wv, val_b, nullptr, nullptr, nullptr, av16, 80000, 64, 0);
  nf_kernel<<<dim3(8000), dim3(64), 0, stream>>>(av16, wbuf, oape16, avln_g,
                                                 avln_b, nf16);
  g64(nf16, Wl1, l1_b, l1g, l1lb, nullptr, t1_16, 80000, 64, F_GELU | F_LN);
  g64(t1_16, Wl2, l2_b, avlinln_g, avlinln_b, nf16, nff16, 80000, 64,
      F_RES | F_LN);

  // ---- enemy branch ----
  g64(es16, Wee, eeb, eeg, eelb, nullptr, ee16, 8800, 128, F_GELU | F_LN);
  g64(se16, Wqe, qe_b, nullptr, nullptr, nullptr, qe16, 8000, 64, 0);
  g64(ee16, Wke, ke_b, nullptr, nullptr, nullptr, ke16, 8800, 64, 0);
  g64(ee16, Wve, ve_b, nullptr, nullptr, nullptr, ve16, 8800, 64, 0);
  enemy_kernel<<<dim3(800), dim3(64), 0, stream>>>(qe16, ke16, ve16, ee16, se16,
                                                   aveln_g, aveln_b, aggb16);
  g64(aggb16, We1, e1_b, e1g, e1lb, nullptr, t2_16, 8000, 64, F_GELU | F_LN);
  g64(t2_16, We2, e2_b, avelnln_g, avelnln_b, aggb16, agg16, 8000, 64,
      F_RES | F_LN);

  // ---- common + GRU input ----
  concat_kernel<<<dim3(60000), dim3(256), 0, stream>>>(se16, agg16, nff16,
                                                       curin16);
  g64(curin16, Wcom, com_b, nullptr, nullptr, nullptr, curx16, 80000, 192,
      F_GELU | F_PERM);
  gemm192_kernel<<<dim3(1250), dim3(128), 0, stream>>>(curx16, Wih, bih, gi16,
                                                       80000);

  // ---- fused GRU + value head (+ hT) ----
  gru_kernel<<<dim3(100), dim3(32), 0, stream>>>(gi16, Whh, bhh_, h0, vg, vbp,
                                                 vw, vbias, outv);
}
